// CoAttention_4913442586768
// MI455X (gfx1250) — matrix operand order repaired and checked by reading, not compiled, not run on hardware
//
#include <hip/hip_runtime.h>
#include <stdint.h>

// ---------------------------------------------------------------------------
// MI455X (gfx1250) implementation.
//
// Roofline: ~125 GFLOP of dense GEMM (fft2 rewritten as DFT-matrix products,
// the bnm einsum, and the two mobius_matvec matmuls) vs <1 GB of HBM traffic
// at 23.3 TB/s -> matrix-core bound. gfx1250's bf16 WMMA (16x16x32) gives 8x
// the K-depth per issue of the f32 shape (16x16x4), and bf16 keeps the f32
// exponent needed by the O(1e2) DFT partial sums, so every GEMM operand is
// bf16 with f32 accumulation; all row-wise hyperbolic math stays f32.
//
// GEMM data path: double-buffered LDS pipeline -- tile t+1 streams in via
// global_load_async_to_lds_b128 (ASYNCcnt, in-order completion) while tile t
// is consumed via ds_load_b128 + v_wmma_f32_16x16x32_bf16. s_wait_asynccnt
// is lowered to the statically known in-flight count of the next tile.
// ---------------------------------------------------------------------------

#define BB    32
#define NSQ   1024      // NS == NC
#define DD    256
#define KK    128
#define DP1   257
#define KP1   129
#define DPAD  288       // 257 padded to multiple of 32 (WMMA K step)
#define KPAD  160       // 129 padded to multiple of 32

typedef __attribute__((ext_vector_type(16))) __bf16 bf16x16;
typedef __attribute__((ext_vector_type(8)))  float  f32x8;

union Frag { bf16x16 v; uint32_t w[8]; };

// ---- CDNA5 async global->LDS copy (CDNA5 ISA) -------------
#define USE_ASYNC_LDS 1
#if USE_ASYNC_LDS
#define ASYNC_B128(ldsptr, gptr)                                              \
    asm volatile("global_load_async_to_lds_b128 %0, %1, off"                  \
                 :: "v"((uint32_t)(uintptr_t)(ldsptr)), "v"(gptr) : "memory")
#define ASYNC_WAIT_N(n)                                                       \
    asm volatile("s_wait_asynccnt %0" :: "i"(n) : "memory")
#else
#define ASYNC_B128(ldsptr, gptr)                                              \
    do { *(uint4*)(ldsptr) = *(const uint4*)(gptr); } while (0)
#define ASYNC_WAIT_N(n) do {} while (0)
#endif

__device__ __forceinline__ float bf2f(uint16_t h) {
    union { uint32_t u; float f; } x; x.u = ((uint32_t)h) << 16; return x.f;
}
__device__ __forceinline__ uint16_t f2bf(float f) {
    union { float f; uint32_t u; } x; x.f = f;
    uint32_t u = x.u;
    return (uint16_t)((u + 0x7FFFu + ((u >> 16) & 1u)) >> 16);  // RNE
}
__device__ __forceinline__ float artanh_f(float x) {
    x = fminf(fmaxf(x, -1.f + 1e-5f), 1.f - 1e-5f);
    return 0.5f * (log1pf(x) - log1pf(-x));
}
__device__ __forceinline__ float gelu_f(float x) {
    return 0.5f * x * (1.f + erff(x * 0.70710678118654752f));
}
__device__ __forceinline__ float block_sum(float v, float* sb) {
    int tid = threadIdx.x;
    sb[tid] = v; __syncthreads();
    for (int s = blockDim.x >> 1; s > 0; s >>= 1) {
        if (tid < s) sb[tid] += sb[tid + s];
        __syncthreads();
    }
    float r = sb[0]; __syncthreads();
    return r;
}
__device__ __forceinline__ float block_max(float v, float* sb) {
    int tid = threadIdx.x;
    sb[tid] = v; __syncthreads();
    for (int s = blockDim.x >> 1; s > 0; s >>= 1) {
        if (tid < s) sb[tid] = fmaxf(sb[tid], sb[tid + s]);
        __syncthreads();
    }
    float r = sb[0]; __syncthreads();
    return r;
}

// ---------------------------------------------------------------------------
// WMMA bf16 GEMM.  C[m,n] = alpha * sum_k Aop[m,k]*Bop[k,n] (+ beta*C + bias)
//   TRANSA=0: A is MxK row-major.   TRANSA=1: A is KxM row-major (A^T GEMM).
//   TRANSB=0: B is KxN row-major.   TRANSB=1: B is NxK row-major (dot rows).
//   OUTBF=1 : write bf16 (beta must be 0), else f32 with optional beta.
// Caller contract: M % 64 == 0, Kd % 32 == 0; N ragged only when TRANSB==1
// (N % 8 == 0); when TRANSB==0, N % 128 == 0.
// Block: 256 threads = 8 wave32 waves, tile 64x128, K-step 32, double-
// buffered LDS; each wave owns a 2x2 grid of 16x16 f32 accumulators.
// ---------------------------------------------------------------------------
#define TM 64
#define TN 128
#define TK 32
#define LSTR 40   // LDS row stride (elements): 80B, multiple of 16B

template<int TRANSA, int TRANSB, int OUTBF>
__global__ __launch_bounds__(256) void gemm_bf16_wmma(
    const uint16_t* __restrict__ A, int lda, long long sA,
    const uint16_t* __restrict__ Bm, int ldb, long long sB,
    void* __restrict__ Cp, int ldc, long long sC,
    int M, int N, int Kd, float alpha, float beta,
    const float* __restrict__ bias)
{
    __shared__ __align__(16) uint16_t lA[2][TM][LSTR];
    __shared__ __align__(16) uint16_t lB[2][TN][LSTR];

    const int bz = blockIdx.z;
    A  += sA * bz;
    Bm += sB * bz;
    float*    Cf = (float*)Cp    + sC * bz;
    uint16_t* Cb = (uint16_t*)Cp + sC * bz;

    const int m0  = blockIdx.y * TM;
    const int n0  = blockIdx.x * TN;
    const int tid = threadIdx.x;
    const int lane = tid & 31;
    const int wave = tid >> 5;
    const int wm = wave >> 2;   // 0..1
    const int wn = wave & 3;    // 0..3
    const int hk  = lane >> 4;  // K-half selector per ISA layout
    const int l15 = lane & 15;
    const bool nfull = (n0 + TN <= N);

    // Per-thread async issues per tile (static -> s_wait_asynccnt immediate).
    constexpr int NPEND = (TRANSA == 0 ? 1 : 0) + (TRANSB == 1 ? 2 : 0);

    f32x8 acc[2][2];
    const f32x8 zero = {0.f, 0.f, 0.f, 0.f, 0.f, 0.f, 0.f, 0.f};
    acc[0][0] = zero; acc[0][1] = zero; acc[1][0] = zero; acc[1][1] = zero;

    auto stageA = [&](int buf, int kk) {
        if (TRANSA == 0) {
            const int r = tid >> 2;            // 0..63
            const int c = (tid & 3) * 8;       // 0,8,16,24
            const uint16_t* gp = &A[(long long)(m0 + r) * lda + kk + c];
            ASYNC_B128(&lA[buf][r][c], gp);
            if (kk + 2 * TK < Kd)
                __builtin_prefetch(gp + 2 * TK, 0, 1);   // warm L2 two panels out
        } else {
            // A is KxM: vector-load 8 along m, scatter-transpose into lA[m][k]
            const int kr = tid >> 3;           // 0..31
            const int mc = (tid & 7) * 8;      // 0..56
            const uint4 va = *(const uint4*)&A[(long long)(kk + kr) * lda + m0 + mc];
            const uint16_t* h = (const uint16_t*)&va;
#pragma unroll
            for (int i = 0; i < 8; ++i) lA[buf][mc + i][kr] = h[i];
        }
    };
    auto stageB = [&](int buf, int kk) {
        if (TRANSB == 1) {
#pragma unroll
            for (int p = 0; p < 2; ++p) {
                const int t = tid + p * 256;
                const int r = t >> 2;          // 0..127
                const int c = (t & 3) * 8;
                const int gn = n0 + r;
                // Clamp keeps issue count uniform; pad rows are zeroed after
                // the wait so the static s_wait_asynccnt threshold stays valid.
                const int gcl = (gn < N) ? gn : (N - 1);
                ASYNC_B128(&lB[buf][r][c], &Bm[(long long)gcl * ldb + kk + c]);
            }
        } else {
            // B is KxN (N % 128 == 0): vector-load 8 along n, scatter-transpose
#pragma unroll
            for (int p = 0; p < 2; ++p) {
                const int t = tid + p * 256;
                const int kr = t >> 4;             // 0..31
                const int nc = (t & 15) * 8;       // 0..120
                const uint4 vb = *(const uint4*)&Bm[(long long)(kk + kr) * ldb + n0 + nc];
                const uint16_t* h = (const uint16_t*)&vb;
#pragma unroll
                for (int i = 0; i < 8; ++i) lB[buf][nc + i][kr] = h[i];
            }
        }
    };
    auto zeroPadB = [&](int buf) {
        if (TRANSB == 1 && !nfull) {
#pragma unroll
            for (int p = 0; p < 2; ++p) {
                const int t = tid + p * 256;
                const int r = t >> 2;
                const int c = (t & 3) * 8;
                if (n0 + r >= N) {
                    const uint4 z4 = {0u, 0u, 0u, 0u};
                    *(uint4*)&lB[buf][r][c] = z4;
                }
            }
        }
    };

    // ---- pipelined main loop: tile t+1 streams in while tile t computes ----
    stageA(0, 0);
    stageB(0, 0);
    const int T = Kd / TK;
    for (int t = 0; t < T; ++t) {
        const int cur = t & 1;
        const int kk = t * TK;
        if (t + 1 < T) {
            stageA(cur ^ 1, kk + TK);
            stageB(cur ^ 1, kk + TK);
            ASYNC_WAIT_N(NPEND);   // in-order completion -> tile t resident
        } else {
            ASYNC_WAIT_N(0);
        }
        zeroPadB(cur);
        __syncthreads();

        // ---- fragments per ISA wave32 layouts ----
        Frag fa[2], fb[2];
#pragma unroll
        for (int i = 0; i < 2; ++i) {
            const int mrow = wm * 32 + i * 16 + l15;
            // A 16x32: lane<16 holds K[0..7],[16..23]; lane>=16 K[8..15],[24..31]
            const uint32_t* p0 = (const uint32_t*)&lA[cur][mrow][hk * 8];
            const uint32_t* p1 = (const uint32_t*)&lA[cur][mrow][16 + hk * 8];
#pragma unroll
            for (int q = 0; q < 4; ++q) { fa[i].w[q] = p0[q]; fa[i].w[4 + q] = p1[q]; }
        }
#pragma unroll
        for (int j = 0; j < 2; ++j) {
            const int nrow = wn * 32 + j * 16 + l15;
            // B 32x16: lane<16 holds K[0..7],[16..23]; lane>=16 K[8..15],[24..31]
            const uint32_t* p0 = (const uint32_t*)&lB[cur][nrow][hk * 8];
            const uint32_t* p1 = (const uint32_t*)&lB[cur][nrow][16 + hk * 8];
#pragma unroll
            for (int q = 0; q < 4; ++q) { fb[j].w[q] = p0[q]; fb[j].w[4 + q] = p1[q]; }
        }
#pragma unroll
        for (int i = 0; i < 2; ++i)
#pragma unroll
            for (int j = 0; j < 2; ++j)
                acc[i][j] = __builtin_amdgcn_wmma_f32_16x16x32_bf16(
                    false, fa[i].v, false, fb[j].v, (short)0, acc[i][j],
                    false, false);
        __syncthreads();
    }

    // ---- epilogue: element e -> m = base + 8*hk + e, n = base + l15 ----
#pragma unroll
    for (int i = 0; i < 2; ++i) {
#pragma unroll
        for (int j = 0; j < 2; ++j) {
            const int nn = n0 + wn * 32 + j * 16 + l15;
            if (!nfull && nn >= N) continue;
#pragma unroll
            for (int e = 0; e < 8; ++e) {
                const int mm = m0 + wm * 32 + i * 16 + hk * 8 + e;
                float v = alpha * acc[i][j][e];
                if (bias) v += bias[nn];
                const long long idx = (long long)mm * ldc + nn;
                if (OUTBF) {
                    Cb[idx] = f2bf(v);
                } else {
                    if (beta != 0.f) v += beta * Cf[idx];
                    Cf[idx] = v;
                }
            }
        }
    }
}

// ---------------------------------------------------------------------------
// Elementwise / row-wise kernels
// ---------------------------------------------------------------------------
__global__ void fill_dft_k(uint16_t* C, uint16_t* S, int n) {
    int idx = blockIdx.x * blockDim.x + threadIdx.x;
    if (idx >= n * n) return;
    int j = idx / n, k = idx % n;
    long long r = ((long long)j * (long long)k) % n;
    float th = (float)(6.283185307179586 * (double)r / (double)n);
    C[idx] = f2bf(cosf(th));
    S[idx] = f2bf(sinf(th));
}

__global__ void cast_bf16_k(const float* __restrict__ x, uint16_t* __restrict__ o, long long n) {
    long long i = (long long)blockIdx.x * blockDim.x + threadIdx.x;
    if (i < n) o[i] = f2bf(x[i]);
}

__global__ void plogmap0_k(const float* __restrict__ x, uint16_t* __restrict__ o) {
    __shared__ float sb[256];
    long long row = blockIdx.x;
    float v = x[row * DD + threadIdx.x];
    float ss = block_sum(v * v, sb);
    float yn = sqrtf(fmaxf(ss, 1e-15f));
    float coef = artanh_f(yn) / yn;
    o[row * DD + threadIdx.x] = f2bf(coef * v);
}

__global__ void euclid_to_lorentz_k(const float* __restrict__ x, uint16_t* __restrict__ L) {
    __shared__ float sb[256];
    long long row = blockIdx.x;
    int tid = threadIdx.x;
    float v = x[row * DD + tid];
    float ss = block_sum(v * v, sb);
    float raw = sqrtf(fmaxf(ss, 1e-15f));
    float xn = raw + 1e-5f;                       // EPS
    float s = fminf(1.f, 2.0f / xn);              // CLIP_R
    float vn = s * raw;
    float sc = sinhf(vn) / vn * s;
    uint16_t* r = L + row * DPAD;
    r[1 + tid] = f2bf(sc * v);
    if (tid == 0) r[0] = f2bf(coshf(vn));
    if (tid < DPAD - DP1) r[DP1 + tid] = 0;       // zero pad
}

__global__ void pad_w_k(const float* __restrict__ W, uint16_t* __restrict__ o,
                        int rows, int cols, int ldo, int orows) {
    int idx = blockIdx.x * blockDim.x + threadIdx.x;
    if (idx >= orows * ldo) return;
    int r = idx / ldo, c = idx % ldo;
    o[idx] = (r < rows && c < cols) ? f2bf(W[(long long)r * cols + c]) : (uint16_t)0;
}

__global__ void pad_b_k(const float* __restrict__ b, float* __restrict__ o, int n, int npad) {
    int idx = blockIdx.x * blockDim.x + threadIdx.x;
    if (idx < npad) o[idx] = (idx < n) ? b[idx] : 0.f;
}

// lorentz_linear tail: recompute time from space, emit bf16 padded row (Wl path)
__global__ void lorentz_time_k(const float* __restrict__ Y, uint16_t* __restrict__ o) {
    __shared__ float sb[256];
    long long row = blockIdx.x;
    int tid = threadIdx.x;
    float v = Y[row * DPAD + 1 + tid];
    float ss = block_sum(v * v, sb);
    float time = sqrtf(ss + 1.f);
    uint16_t* r = o + row * DPAD;
    r[1 + tid] = f2bf(v);
    if (tid == 0) r[0] = f2bf(time);
    if (tid < DPAD - DP1) r[DP1 + tid] = 0;
}

// lorentz_act over 1024-wide rows + row norms for mobius_matvec
__global__ void lorentz_act_k(const float* __restrict__ E, uint16_t* __restrict__ Lact,
                              float* __restrict__ xnorm) {
    __shared__ float sb[256];
    long long row = blockIdx.x;
    long long base = row * (long long)NSQ;
    int tid = threadIdx.x;
    float g[4]; float ss = 0.f;
#pragma unroll
    for (int p = 0; p < 4; ++p) {
        int j = tid + p * 256;
        float x = E[base + j];
        float gg = (j == 0) ? 0.f : gelu_f(x);
        g[p] = gg; ss += gg * gg;
    }
    float tot = block_sum(ss, sb);
    float time = sqrtf(tot + 1.f);
#pragma unroll
    for (int p = 0; p < 4; ++p) {
        int j = tid + p * 256;
        Lact[base + j] = f2bf((j == 0) ? time : g[p]);
    }
    if (tid == 0) xnorm[row] = sqrtf(2.f * tot + 1.f);  // ||[time, space]||
}

__global__ void colnorm_k(const uint16_t* __restrict__ Lact, float* __restrict__ xnorm) {
    __shared__ float sb[256];
    int m = blockIdx.x, b = blockIdx.y;
    const uint16_t* base = Lact + (long long)b * NSQ * NSQ + m;
    float ss = 0.f;
    for (int n = threadIdx.x; n < NSQ; n += 256) {
        float v = bf2f(base[(long long)n * NSQ]);
        ss += v * v;
    }
    float tot = block_sum(ss, sb);
    if (threadIdx.x == 0) xnorm[(long long)b * NSQ + m] = sqrtf(fmaxf(tot, 1e-15f));
}

// lorentz_linear(Ws/Wc) tail + lorentz_to_poincare (128 space dims)
__global__ void time_poincare_k(const float* __restrict__ Y, float* __restrict__ Pf,
                                uint16_t* __restrict__ Pb) {
    __shared__ float sb[256];
    long long row = blockIdx.x;
    int tid = threadIdx.x;
    float v = Y[row * KPAD + 1 + tid];
    float ss = block_sum(v * v, sb);
    float time = sqrtf(ss + 1.f);
    float p = v / (time + 1.f);
    Pf[row * KK + tid] = p;
    Pb[row * KK + tid] = f2bf(p);
}

// mobius_matvec scaling: out = tanh(|mx|/xn * artanh(xn)) * mx/|mx|
__global__ void mvec_scale_k(const float* __restrict__ MX, const float* __restrict__ xn_a,
                             float* __restrict__ O) {
    __shared__ float sb[256];
    long long row = blockIdx.x;
    int tid = threadIdx.x;
    float v = MX[row * KK + tid];
    float ss = block_sum(v * v, sb);
    float mxn = sqrtf(fmaxf(ss, 1e-15f));
    float xn = xn_a[row];
    float t = tanhf(mxn / xn * artanh_f(xn));
    O[row * KK + tid] = t * v / mxn;
}

// mobius_add -> p_logmap0 -> gelu -> p_expmap0 over 128-dim rows
__global__ void hs_final_k(const float* __restrict__ Xa, const float* __restrict__ Yb,
                           float* __restrict__ Hs) {
    __shared__ float sb[256];
    long long row = blockIdx.x;
    int tid = threadIdx.x;
    float x = Xa[row * KK + tid], y = Yb[row * KK + tid];
    float xy = block_sum(x * y, sb);
    float x2 = block_sum(x * x, sb);
    float y2 = block_sum(y * y, sb);
    float den = fmaxf(1.f + 2.f * xy + x2 * y2, 1e-15f);
    float h = ((1.f + 2.f * xy + y2) * x + (1.f - x2) * y) / den;
    float hn = sqrtf(fmaxf(block_sum(h * h, sb), 1e-15f));
    float u = artanh_f(hn) / hn * h;
    float g = gelu_f(u);
    float gn = sqrtf(fmaxf(block_sum(g * g, sb), 1e-15f));
    Hs[row * KK + tid] = tanhf(gn) / gn * g;
}

// same pipeline over 1024-dim rows: x[n]=Hc_a[b,n,k], y[n]=Hc_b[b,n,k]
__global__ void hc_final_k(const float* __restrict__ Xa, const float* __restrict__ Yb,
                           float* __restrict__ Hc) {
    __shared__ float sb[256];
    int k = blockIdx.x, b = blockIdx.y, tid = threadIdx.x;
    float x[4], y[4];
#pragma unroll
    for (int p = 0; p < 4; ++p) {
        int n = tid + p * 256;
        long long idx = ((long long)b * NSQ + n) * KK + k;
        x[p] = Xa[idx]; y[p] = Yb[idx];
    }
    float lxy = 0.f, lx2 = 0.f, ly2 = 0.f;
#pragma unroll
    for (int p = 0; p < 4; ++p) { lxy += x[p] * y[p]; lx2 += x[p] * x[p]; ly2 += y[p] * y[p]; }
    float xy = block_sum(lxy, sb);
    float x2 = block_sum(lx2, sb);
    float y2 = block_sum(ly2, sb);
    float den = fmaxf(1.f + 2.f * xy + x2 * y2, 1e-15f);
    float h[4]; float lh2 = 0.f;
#pragma unroll
    for (int p = 0; p < 4; ++p) {
        h[p] = ((1.f + 2.f * xy + y2) * x[p] + (1.f - x2) * y[p]) / den;
        lh2 += h[p] * h[p];
    }
    float hn = sqrtf(fmaxf(block_sum(lh2, sb), 1e-15f));
    float coef = artanh_f(hn) / hn;
    float g[4]; float lg2 = 0.f;
#pragma unroll
    for (int p = 0; p < 4; ++p) { g[p] = gelu_f(coef * h[p]); lg2 += g[p] * g[p]; }
    float gn = sqrtf(fmaxf(block_sum(lg2, sb), 1e-15f));
    float sc = tanhf(gn) / gn;
#pragma unroll
    for (int p = 0; p < 4; ++p)
        Hc[((long long)b * KK + k) * NSQ + tid + p * 256] = sc * g[p];
}

__global__ void score_s_k(const float* __restrict__ Hs, const float* __restrict__ wh,
                          float* __restrict__ sc) {
    int idx = blockIdx.x * blockDim.x + threadIdx.x;
    if (idx >= BB * NSQ) return;
    const float* r = Hs + (long long)idx * KK;
    float mx = 0.f, ss = 0.f;
    for (int j = 0; j < KK; ++j) { float v = r[j]; mx += v * wh[j]; ss += v * v; }
    float xn = sqrtf(fmaxf(ss, 1e-15f));
    float mxn = fmaxf(fabsf(mx), 1e-15f);
    sc[idx] = tanhf(mxn / xn * artanh_f(xn)) * mx / mxn;
}

__global__ void score_c_k(const float* __restrict__ Hc, const float* __restrict__ wh,
                          float* __restrict__ sc) {
    int idx = blockIdx.x * blockDim.x + threadIdx.x;
    if (idx >= BB * NSQ) return;
    int b = idx >> 10, n = idx & 1023;
    float mx = 0.f, ss = 0.f;
    for (int j = 0; j < KK; ++j) {
        float v = Hc[((long long)b * KK + j) * NSQ + n];
        mx += v * wh[j]; ss += v * v;
    }
    float xn = sqrtf(fmaxf(ss, 1e-15f));
    float mxn = fmaxf(fabsf(mx), 1e-15f);
    sc[idx] = tanhf(mxn / xn * artanh_f(xn)) * mx / mxn;
}

__global__ void softmax_k(const float* __restrict__ sc, float* __restrict__ o) {
    __shared__ float sb[256];
    int b = blockIdx.x, tid = threadIdx.x;
    float m = -1e30f;
    for (int n = tid; n < NSQ; n += 256) m = fmaxf(m, sc[b * NSQ + n]);
    m = block_max(m, sb);
    float s = 0.f;
    for (int n = tid; n < NSQ; n += 256) s += expf(sc[b * NSQ + n] - m);
    s = block_sum(s, sb);
    for (int n = tid; n < NSQ; n += 256) o[b * NSQ + n] = expf(sc[b * NSQ + n] - m) / s;
}

__global__ void centroid_k(const uint16_t* __restrict__ L, const float* __restrict__ Aw,
                           float* __restrict__ co) {
    __shared__ float avg[DP1];
    __shared__ float sb[256];
    int b = blockIdx.x, tid = threadIdx.x;
    for (int d = tid; d < DP1; d += 256) {
        float acc = 0.f;
        const uint16_t* base = L + ((long long)b * NSQ) * DPAD + d;
        const float* aw = Aw + b * NSQ;
        for (int n = 0; n < NSQ; ++n) acc += aw[n] * bf2f(base[(long long)n * DPAD]);
        avg[d] = acc;
    }
    __syncthreads();
    float part = 0.f;
    for (int d = tid; d < DP1; d += 256) {
        float v = avg[d];
        part += (d == 0) ? -v * v : v * v;
    }
    float inner = block_sum(part, sb);
    float den = sqrtf(fmaxf(fabsf(inner), 1e-8f));
    for (int d = tid; d < DP1; d += 256) co[(long long)b * DP1 + d] = avg[d] / den;
}

__global__ void concat_final_k(const float* __restrict__ co_s, const float* __restrict__ co_c,
                               float* __restrict__ out) {
    __shared__ float z[512];
    __shared__ float sb[256];
    int b = blockIdx.x, tid = threadIdx.x;
    const float* cs = co_s + (long long)b * DP1;
    const float* cc = co_c + (long long)b * DP1;
    float v1 = cs[1 + tid];
    float n1 = sqrtf(fmaxf(block_sum(v1 * v1, sb), 1e-15f));
    float t1 = fmaxf(cs[0], 1.f + 1e-7f);
    z[tid] = acoshf(t1) / n1 * v1;
    float v2 = cc[1 + tid];
    float n2 = sqrtf(fmaxf(block_sum(v2 * v2, sb), 1e-15f));
    float t2 = fmaxf(cc[0], 1.f + 1e-7f);
    z[256 + tid] = acoshf(t2) / n2 * v2;
    __syncthreads();
    float zz = z[tid] * z[tid] + z[256 + tid] * z[256 + tid];
    float vn = sqrtf(fmaxf(block_sum(zz, sb), 1e-15f));
    float sc = sinhf(vn) / vn;
    float* o = out + (long long)b * 513;
    o[1 + tid] = sc * z[tid];
    o[1 + 256 + tid] = sc * z[256 + tid];
    if (tid == 0) o[0] = coshf(vn);
}

// ---------------------------------------------------------------------------
extern "C" void kernel_launch(void* const* d_in, const int* in_sizes, int n_in,
                              void* d_out, int out_size, void* d_ws, size_t ws_size,
                              hipStream_t stream) {
    (void)in_sizes; (void)n_in; (void)out_size;
    const float* sentence = (const float*)d_in[0];
    const float* comment  = (const float*)d_in[1];
    const float* WlW = (const float*)d_in[2];
    const float* Wlb = (const float*)d_in[3];
    const float* WcW = (const float*)d_in[4];
    const float* Wcb = (const float*)d_in[5];
    const float* WsW = (const float*)d_in[6];
    const float* Wsb = (const float*)d_in[7];
    const float* whs = (const float*)d_in[8];
    const float* whc = (const float*)d_in[9];
    float* out = (float*)d_out;

    char* base = (char*)d_ws;
    size_t off = 0;
    auto alloc = [&](size_t bytes) -> char* {
        size_t p = (off + 255) & ~(size_t)255;
        off = p + bytes;
        return base + p;
    };
    const long long NR = (long long)BB * NSQ;  // 32768 rows

    uint16_t* dftC_N = (uint16_t*)alloc((size_t)NSQ * NSQ * 2);
    uint16_t* dftS_N = (uint16_t*)alloc((size_t)NSQ * NSQ * 2);
    uint16_t* dftC_D = (uint16_t*)alloc((size_t)DD * DD * 2);
    uint16_t* dftS_D = (uint16_t*)alloc((size_t)DD * DD * 2);
    uint16_t* Xbf = (uint16_t*)alloc((size_t)NR * DD * 2);
    uint16_t* Tc  = (uint16_t*)alloc((size_t)NR * DD * 2);
    uint16_t* Ts  = (uint16_t*)alloc((size_t)NR * DD * 2);
    float*    sF  = (float*)alloc((size_t)NR * DD * 4);
    float*    cF  = (float*)alloc((size_t)NR * DD * 4);
    uint16_t* Ls  = (uint16_t*)alloc((size_t)NR * DPAD * 2);
    uint16_t* Lc  = (uint16_t*)alloc((size_t)NR * DPAD * 2);
    uint16_t* Lw  = (uint16_t*)alloc((size_t)NR * DPAD * 2);
    uint16_t* Wlp = (uint16_t*)alloc((size_t)DPAD * DPAD * 2);
    uint16_t* Wsp = (uint16_t*)alloc((size_t)KPAD * DPAD * 2);
    uint16_t* Wcp = (uint16_t*)alloc((size_t)KPAD * DPAD * 2);
    float* blp = (float*)alloc(DPAD * 4);
    float* bsp = (float*)alloc(KPAD * 4);
    float* bcp = (float*)alloc(KPAD * 4);
    float* Ybig = (float*)alloc((size_t)NR * DPAD * 4);
    float* Ysm  = (float*)alloc((size_t)NR * KPAD * 4);
    float* E    = (float*)alloc((size_t)NR * NSQ * 4);
    uint16_t* Lact = (uint16_t*)alloc((size_t)NR * NSQ * 2);
    float* xnorm_s = (float*)alloc((size_t)NR * 4);
    float* xnorm_c = (float*)alloc((size_t)NR * 4);
    float* HsaF = (float*)alloc((size_t)NR * KK * 4);
    uint16_t* HsaB = (uint16_t*)alloc((size_t)NR * KK * 2);
    float* HcaF = (float*)alloc((size_t)NR * KK * 4);
    uint16_t* HcaB = (uint16_t*)alloc((size_t)NR * KK * 2);
    float* MX   = (float*)alloc((size_t)NR * KK * 4);
    float* HsbF = (float*)alloc((size_t)NR * KK * 4);
    float* HcbF = (float*)alloc((size_t)NR * KK * 4);
    float* Hs   = (float*)alloc((size_t)NR * KK * 4);
    float* Hc   = (float*)alloc((size_t)NR * KK * 4);
    float* scs  = (float*)alloc((size_t)NR * 4);
    float* scc  = (float*)alloc((size_t)NR * 4);
    float* co_s = (float*)alloc((size_t)BB * DP1 * 4);
    float* co_c = (float*)alloc((size_t)BB * DP1 * 4);
    if (off > ws_size) return;

    float* outCo = out;
    float* outAs = out + (size_t)BB * 513;
    float* outAc = outAs + (size_t)BB * NSQ;

    const dim3 blk(256);
    const long long sXD = (long long)NSQ * DD;
    const long long sXP = (long long)NSQ * DPAD;

    // DFT matrices (fft2 -> Re = C_N X C_D - S_N X S_D)
    fill_dft_k<<<(NSQ * NSQ + 255) / 256, blk, 0, stream>>>(dftC_N, dftS_N, NSQ);
    fill_dft_k<<<(DD * DD + 255) / 256, blk, 0, stream>>>(dftC_D, dftS_D, DD);

    // ---- sentence: fft2 -> Lorentz ----
    cast_bf16_k<<<(int)((NR * DD + 255) / 256), blk, 0, stream>>>(sentence, Xbf, NR * DD);
    {
        dim3 g(2, 16, BB);
        gemm_bf16_wmma<0,0,1><<<g, blk, 0, stream>>>(dftC_N, NSQ, 0, Xbf, DD, sXD, Tc, DD, sXD, NSQ, DD, NSQ, 1.f, 0.f, nullptr);
        gemm_bf16_wmma<0,0,1><<<g, blk, 0, stream>>>(dftS_N, NSQ, 0, Xbf, DD, sXD, Ts, DD, sXD, NSQ, DD, NSQ, 1.f, 0.f, nullptr);
        gemm_bf16_wmma<0,0,0><<<g, blk, 0, stream>>>(Tc, DD, sXD, dftC_D, DD, 0, sF, DD, sXD, NSQ, DD, DD,  1.f, 0.f, nullptr);
        gemm_bf16_wmma<0,0,0><<<g, blk, 0, stream>>>(Ts, DD, sXD, dftS_D, DD, 0, sF, DD, sXD, NSQ, DD, DD, -1.f, 1.f, nullptr);
    }
    euclid_to_lorentz_k<<<(int)NR, blk, 0, stream>>>(sF, Ls);

    // ---- comment: p_logmap0 -> fft2 -> Lorentz ----
    plogmap0_k<<<(int)NR, blk, 0, stream>>>(comment, Xbf);
    {
        dim3 g(2, 16, BB);
        gemm_bf16_wmma<0,0,1><<<g, blk, 0, stream>>>(dftC_N, NSQ, 0, Xbf, DD, sXD, Tc, DD, sXD, NSQ, DD, NSQ, 1.f, 0.f, nullptr);
        gemm_bf16_wmma<0,0,1><<<g, blk, 0, stream>>>(dftS_N, NSQ, 0, Xbf, DD, sXD, Ts, DD, sXD, NSQ, DD, NSQ, 1.f, 0.f, nullptr);
        gemm_bf16_wmma<0,0,0><<<g, blk, 0, stream>>>(Tc, DD, sXD, dftC_D, DD, 0, cF, DD, sXD, NSQ, DD, DD,  1.f, 0.f, nullptr);
        gemm_bf16_wmma<0,0,0><<<g, blk, 0, stream>>>(Ts, DD, sXD, dftS_D, DD, 0, cF, DD, sXD, NSQ, DD, DD, -1.f, 1.f, nullptr);
    }
    euclid_to_lorentz_k<<<(int)NR, blk, 0, stream>>>(cF, Lc);

    // ---- weights: pad+cast ----
    pad_w_k<<<(DPAD * DPAD + 255) / 256, blk, 0, stream>>>(WlW, Wlp, DP1, DP1, DPAD, DPAD);
    pad_w_k<<<(KPAD * DPAD + 255) / 256, blk, 0, stream>>>(WsW, Wsp, KP1, DP1, DPAD, KPAD);
    pad_w_k<<<(KPAD * DPAD + 255) / 256, blk, 0, stream>>>(WcW, Wcp, KP1, DP1, DPAD, KPAD);
    pad_b_k<<<2, blk, 0, stream>>>(Wlb, blp, DP1, DPAD);
    pad_b_k<<<1, blk, 0, stream>>>(Wsb, bsp, KP1, KPAD);
    pad_b_k<<<1, blk, 0, stream>>>(Wcb, bcp, KP1, KPAD);

    // ---- L = lorentz_linear(Lc, Wl); E = einsum(Ls, L); lorentz_act ----
    {
        dim3 g((DPAD + TN - 1) / TN, (int)(NR / TM), 1);
        gemm_bf16_wmma<0,1,0><<<g, blk, 0, stream>>>(Lc, DPAD, 0, Wlp, DPAD, 0, Ybig, DPAD, 0, (int)NR, DPAD, DPAD, 1.f, 0.f, blp);
    }
    lorentz_time_k<<<(int)NR, blk, 0, stream>>>(Ybig, Lw);
    {
        dim3 g(NSQ / TN, NSQ / TM, BB);
        gemm_bf16_wmma<0,1,0><<<g, blk, 0, stream>>>(Ls, DPAD, sXP, Lw, DPAD, sXP, E, NSQ, (long long)NSQ * NSQ, NSQ, NSQ, DPAD, 1.f, 0.f, nullptr);
    }
    lorentz_act_k<<<(int)NR, blk, 0, stream>>>(E, Lact, xnorm_s);
    colnorm_k<<<dim3(NSQ, BB), blk, 0, stream>>>(Lact, xnorm_c);

    // ---- Hs_a / Hc_a: lorentz_linear + lorentz_to_poincare ----
    {
        dim3 g((KPAD + TN - 1) / TN, (int)(NR / TM), 1);
        gemm_bf16_wmma<0,1,0><<<g, blk, 0, stream>>>(Ls, DPAD, 0, Wsp, DPAD, 0, Ysm, KPAD, 0, (int)NR, KPAD, DPAD, 1.f, 0.f, bsp);
    }
    time_poincare_k<<<(int)NR, 128, 0, stream>>>(Ysm, HsaF, HsaB);
    {
        dim3 g((KPAD + TN - 1) / TN, (int)(NR / TM), 1);
        gemm_bf16_wmma<0,1,0><<<g, blk, 0, stream>>>(Lc, DPAD, 0, Wcp, DPAD, 0, Ysm, KPAD, 0, (int)NR, KPAD, DPAD, 1.f, 0.f, bcp);
    }
    time_poincare_k<<<(int)NR, 128, 0, stream>>>(Ysm, HcaF, HcaB);

    // ---- mobius_matvec matmuls: Lact @ Hc_a and Lact^T @ Hs_a ----
    {
        dim3 g(1, NSQ / TM, BB);
        gemm_bf16_wmma<0,0,0><<<g, blk, 0, stream>>>(Lact, NSQ, (long long)NSQ * NSQ, HcaB, KK, (long long)NSQ * KK, MX, KK, (long long)NSQ * KK, NSQ, KK, NSQ, 1.f, 0.f, nullptr);
    }
    mvec_scale_k<<<(int)NR, 128, 0, stream>>>(MX, xnorm_s, HsbF);
    {
        dim3 g(1, NSQ / TM, BB);
        gemm_bf16_wmma<1,0,0><<<g, blk, 0, stream>>>(Lact, NSQ, (long long)NSQ * NSQ, HsaB, KK, (long long)NSQ * KK, MX, KK, (long long)NSQ * KK, NSQ, KK, NSQ, 1.f, 0.f, nullptr);
    }
    mvec_scale_k<<<(int)NR, 128, 0, stream>>>(MX, xnorm_c, HcbF);

    // ---- mobius_add + log/gelu/exp maps ----
    hs_final_k<<<(int)NR, 128, 0, stream>>>(HsaF, HsbF, Hs);
    hc_final_k<<<dim3(KK, BB), blk, 0, stream>>>(HcaF, HcbF, Hc);

    // ---- attention scores, softmax, centroids, concat ----
    score_s_k<<<(int)((NR + 255) / 256), blk, 0, stream>>>(Hs, whs, scs);
    score_c_k<<<(int)((NR + 255) / 256), blk, 0, stream>>>(Hc, whc, scc);
    softmax_k<<<BB, blk, 0, stream>>>(scs, outAs);
    softmax_k<<<BB, blk, 0, stream>>>(scc, outAc);
    centroid_k<<<BB, blk, 0, stream>>>(Ls, outAs, co_s);
    centroid_k<<<BB, blk, 0, stream>>>(Lc, outAc, co_c);
    concat_final_k<<<BB, blk, 0, stream>>>(co_s, co_c, outCo);
}
